// LineEncoder_2516850835871
// MI455X (gfx1250) — matrix operand order repaired and checked by reading, not compiled, not run on hardware
//
#include <hip/hip_runtime.h>
#include <hip/hip_bf16.h>
#include <math.h>

// ---------------------------------------------------------------------------
// CDNA5 (gfx1250) types + helpers
// ---------------------------------------------------------------------------
typedef __attribute__((ext_vector_type(16))) __bf16 v16bf;
typedef __attribute__((ext_vector_type(8)))  __bf16 v8bf;
typedef __attribute__((ext_vector_type(8)))  float  v8f;

// native f32 -> bf16 convert (gfx1250 has hardware bf16 cvt; RNE)
static __device__ __forceinline__ __bf16 f2bf(float f) { return (__bf16)f; }

#define D_MODEL 256
#define N_HEAD  8
#define DK      32
#define BB      4
#define NN      24
#define LL      48
#define WW      8
#define BN      (BB * NN)          // 96
#define ROWS    (BN * LL)          // 4608
#define NEGBIG  (-1e9f)
#define RSQRT_DK 0.17677669529663687f

// GEMM block tiling: 64(M) x 64(N), 8 waves, each wave = two 16x16 M-tiles
#define BM   64
#define BNT  64
#define BK   32
#define LBS  40                    // lB row stride in bf16 (80B -> 16B-aligned frags)

#define SHUF16(lo, hi) __builtin_shufflevector(lo, hi, 0,1,2,3,4,5,6,7,8,9,10,11,12,13,14,15)

// ---------------------------------------------------------------------------
// Generic bf16-WMMA GEMM:  out = epi(A[MxK] @ W[KxN])
//   epilogue: + bias[col]*rowscale[row/1152]  -> relu? -> + resid
// Double-buffered LDS staging (one barrier per k-step); each wave issues two
// independent v_wmma per k-step (shared B fragment) to hide WMMA hazards.
// Grid = (M/64, N/64). M%64==0, N%64==0, K%32==0 guaranteed by caller.
// ---------------------------------------------------------------------------
__global__ void gemm_wmma_kernel(const float* __restrict__ A,
                                 const float* __restrict__ W,
                                 const float* __restrict__ bias,
                                 const float* __restrict__ resid,
                                 const float* __restrict__ rowscale,
                                 float* __restrict__ out,
                                 int M, int K, int N, int do_relu) {
    __shared__ __bf16 lA[2][BM * BK];      // [m][k], row stride 32 (64B)
    __shared__ __bf16 lB[2][BNT * LBS];    // [n][k], row stride 40 (80B)

    const int tid  = threadIdx.x;
    const int wid  = tid >> 5;
    const int lane = tid & 31;
    const int wn   = wid & 3;              // N tile (0..3)
    const int wmg  = wid >> 2;             // M group (0..1): rows wmg*32 + {0..31}
    const int hi   = lane >> 4;
    const int l15  = lane & 15;
    const int mblk = blockIdx.x * BM;
    const int nblk = blockIdx.y * BNT;

    // staging split (256 threads)
    const int ar = tid >> 2;               // 0..63  A row
    const int ak = (tid & 3) * 8;          // A k group of 8
    const int bk = tid >> 3;               // 0..31  B k row
    const int bn = (tid & 7) * 8;          // B n group of 8

    const float* aln = A + (size_t)(mblk + ar) * K + ak;   // this thread's A chunk
    const float* wln = W + (size_t)nblk + bn;              // base of this thread's W cols

    v8f acc0 = {0.f,0.f,0.f,0.f,0.f,0.f,0.f,0.f};
    v8f acc1 = {0.f,0.f,0.f,0.f,0.f,0.f,0.f,0.f};

    auto stage = [&](int kk, int b) {
        // A tile (64x32 f32 -> bf16), one packed 16B ds_store per thread
        const float4 a0 = *(const float4*)(aln + kk);
        const float4 a1 = *(const float4*)(aln + kk + 4);
        v8bf pk;
        pk[0] = f2bf(a0.x); pk[1] = f2bf(a0.y); pk[2] = f2bf(a0.z); pk[3] = f2bf(a0.w);
        pk[4] = f2bf(a1.x); pk[5] = f2bf(a1.y); pk[6] = f2bf(a1.z); pk[7] = f2bf(a1.w);
        *(v8bf*)(&lA[b][ar * BK + ak]) = pk;
        // B tile (32x64 f32 -> bf16), transposed to [n][k]
        const float* wp = wln + (size_t)(kk + bk) * N;
        const float4 w0 = *(const float4*)(wp);
        const float4 w1 = *(const float4*)(wp + 4);
        lB[b][(bn + 0) * LBS + bk] = f2bf(w0.x);
        lB[b][(bn + 1) * LBS + bk] = f2bf(w0.y);
        lB[b][(bn + 2) * LBS + bk] = f2bf(w0.z);
        lB[b][(bn + 3) * LBS + bk] = f2bf(w0.w);
        lB[b][(bn + 4) * LBS + bk] = f2bf(w1.x);
        lB[b][(bn + 5) * LBS + bk] = f2bf(w1.y);
        lB[b][(bn + 6) * LBS + bk] = f2bf(w1.z);
        lB[b][(bn + 7) * LBS + bk] = f2bf(w1.w);
    };

    stage(0, 0);
    __syncthreads();
    int buf = 0;
    for (int kk = 0; kk < K; kk += BK) {
        if (kk + BK < K) {
            stage(kk + BK, buf ^ 1);                      // overlap with WMMA below
            __builtin_prefetch(aln + kk + 2 * BK, 0, 1);  // L2 prefetch 2 panels ahead
            __builtin_prefetch(wln + (size_t)(kk + 2 * BK + bk) * N, 0, 1);
        }
        // fragments from current buffer: 16B ds reads
        const int mloc = wmg * 32 + l15;
        const v8bf a00 = *(const v8bf*)(&lA[buf][mloc * BK + hi * 8]);
        const v8bf a01 = *(const v8bf*)(&lA[buf][mloc * BK + 16 + hi * 8]);
        const v8bf a10 = *(const v8bf*)(&lA[buf][(mloc + 16) * BK + hi * 8]);
        const v8bf a11 = *(const v8bf*)(&lA[buf][(mloc + 16) * BK + 16 + hi * 8]);
        const int nloc = wn * 16 + l15;
        const v8bf b0 = *(const v8bf*)(&lB[buf][nloc * LBS + hi * 8]);
        const v8bf b1 = *(const v8bf*)(&lB[buf][nloc * LBS + 16 + hi * 8]);
        const v16bf af0 = SHUF16(a00, a01);
        const v16bf af1 = SHUF16(a10, a11);
        const v16bf bfv = SHUF16(b0, b1);
        // two independent WMMAs hide each other's hazard window
        acc0 = __builtin_amdgcn_wmma_f32_16x16x32_bf16(false, af0, false, bfv,
                                                       (short)0, acc0, false, false);
        acc1 = __builtin_amdgcn_wmma_f32_16x16x32_bf16(false, af1, false, bfv,
                                                       (short)0, acc1, false, false);
        __syncthreads();
        buf ^= 1;
    }

    const int bcol = nblk + wn * 16 + l15;
    #pragma unroll
    for (int t = 0; t < 2; ++t) {
        const v8f acc = t ? acc1 : acc0;
        const int rbase = mblk + wmg * 32 + t * 16;
        #pragma unroll
        for (int r = 0; r < 8; ++r) {
            const int row = rbase + r + hi * 8;
            float v = acc[r];
            if (bias) {
                const float sc = rowscale ? rowscale[row / (NN * LL)] : 1.0f;
                v += bias[bcol] * sc;
            }
            if (do_relu) v = fmaxf(v, 0.0f);
            if (resid)   v += resid[(size_t)row * N + bcol];
            out[(size_t)row * N + bcol] = v;
        }
    }
}

// ---------------------------------------------------------------------------
// Char-conv as gathered WMMA GEMM:
//   out[r,o] = sum_{k=(d*8+w)} char_emb[codeChar[r,w], d] * conv_w[o,d,w] + b[o]
// conv_w is [o][d][w] == B[n][k] contiguous per n -> B frag = 4x float4 loads.
// M=4608, K=2048, N=256. Grid (144, 4), block 256.
// ---------------------------------------------------------------------------
__global__ void charconv_wmma_kernel(const int*   __restrict__ codeChar,
                                     const float* __restrict__ char_emb,
                                     const float* __restrict__ conv_w,
                                     const float* __restrict__ conv_b,
                                     float* __restrict__ out) {
    const int K = 2048, N = 256;
    const int wid  = threadIdx.x >> 5;
    const int lane = threadIdx.x & 31;
    const int wm = wid & 1, wn = wid >> 1;
    const int m0 = blockIdx.x * 32 + wm * 16;
    const int n0 = blockIdx.y * 64 + wn * 16;
    const int mrow = lane & 15;
    const int hi   = lane >> 4;
    const int arow = m0 + mrow;      // token row
    const int bcol = n0 + mrow;      // output channel o

    int cidx[8];
    #pragma unroll
    for (int w = 0; w < 8; ++w) cidx[w] = codeChar[arow * 8 + w];
    const float* brow = conv_w + (size_t)bcol * K;     // contiguous in k

    v8f acc = {0.f, 0.f, 0.f, 0.f, 0.f, 0.f, 0.f, 0.f};
    for (int kk = 0; kk < K; kk += 32) {
        v16bf af, bfv;
        // A: embedding gather (8-runs share d, cycle w)
        #pragma unroll
        for (int e = 0; e < 16; ++e) {
            const int ka = kk + hi * 16 + e;
            af[e] = f2bf(char_emb[cidx[ka & 7] * 256 + (ka >> 3)]);
        }
        // B: contiguous 16 floats at conv_w[bcol][kk + hi*16 ..]
        const float4 w0 = *(const float4*)(brow + kk + hi * 16 + 0);
        const float4 w1 = *(const float4*)(brow + kk + hi * 16 + 4);
        const float4 w2 = *(const float4*)(brow + kk + hi * 16 + 8);
        const float4 w3 = *(const float4*)(brow + kk + hi * 16 + 12);
        bfv[0]  = f2bf(w0.x); bfv[1]  = f2bf(w0.y); bfv[2]  = f2bf(w0.z); bfv[3]  = f2bf(w0.w);
        bfv[4]  = f2bf(w1.x); bfv[5]  = f2bf(w1.y); bfv[6]  = f2bf(w1.z); bfv[7]  = f2bf(w1.w);
        bfv[8]  = f2bf(w2.x); bfv[9]  = f2bf(w2.y); bfv[10] = f2bf(w2.z); bfv[11] = f2bf(w2.w);
        bfv[12] = f2bf(w3.x); bfv[13] = f2bf(w3.y); bfv[14] = f2bf(w3.z); bfv[15] = f2bf(w3.w);
        acc = __builtin_amdgcn_wmma_f32_16x16x32_bf16(false, af, false, bfv,
                                                      (short)0, acc, false, false);
    }
    #pragma unroll
    for (int r = 0; r < 8; ++r)
        out[(size_t)(m0 + r + hi * 8) * N + bcol] = acc[r] + conv_b[bcol];
}

// ---------------------------------------------------------------------------
// Token embedding + per-line sinusoidal positional embedding
// ---------------------------------------------------------------------------
__global__ void embed_kernel(const int* __restrict__ code,
                             const float* __restrict__ tok,
                             float* __restrict__ x) {
    const int gid = blockIdx.x * 256 + threadIdx.x;       // over ROWS*256
    const int d = gid & 255, r = gid >> 8;
    const int nline = (r / LL) % NN;
    const int i2 = d & ~1;
    const float inv = __expf(-9.210340371976184f * (float)i2 * (1.0f / 256.0f));
    const float ang = (float)nline * inv;
    const float pe  = (d & 1) ? __cosf(ang) : __sinf(ang);
    x[gid] = tok[(size_t)code[r] * 256 + d] + pe;
}

// ---------------------------------------------------------------------------
// Fused (a = x + charEm) + LayerNorm(a) -> writes both a and h.
// One wave per row, 8 rows per 256-thread block.
// ---------------------------------------------------------------------------
__global__ void add_ln_kernel(const float* __restrict__ x, const float* __restrict__ c,
                              const float* __restrict__ g, const float* __restrict__ b,
                              float* __restrict__ aout, float* __restrict__ hout) {
    const int wid  = threadIdx.x >> 5;
    const int lane = threadIdx.x & 31;
    const int row  = blockIdx.x * 8 + wid;
    const size_t base = (size_t)row * 256;
    float v[8], s = 0.f;
    #pragma unroll
    for (int i = 0; i < 8; ++i) {
        const int cc = lane + i * 32;
        v[i] = x[base + cc] + c[base + cc];
        aout[base + cc] = v[i];
        s += v[i];
    }
    #pragma unroll
    for (int m = 16; m > 0; m >>= 1) s += __shfl_xor(s, m, 32);
    const float mean = s * (1.0f / 256.0f);
    float q = 0.f;
    #pragma unroll
    for (int i = 0; i < 8; ++i) { const float d0 = v[i] - mean; q += d0 * d0; }
    #pragma unroll
    for (int m = 16; m > 0; m >>= 1) q += __shfl_xor(q, m, 32);
    const float inv = rsqrtf(q * (1.0f / 256.0f) + 1e-6f);
    #pragma unroll
    for (int i = 0; i < 8; ++i) {
        const int cc = lane + i * 32;
        hout[base + cc] = g[cc] * (v[i] - mean) * inv + b[cc];
    }
}

// Plain LayerNorm (same layout)
__global__ void ln_kernel(const float* __restrict__ in,
                          const float* __restrict__ g, const float* __restrict__ b,
                          float* __restrict__ out) {
    const int wid  = threadIdx.x >> 5;
    const int lane = threadIdx.x & 31;
    const int row  = blockIdx.x * 8 + wid;
    const float* p = in + (size_t)row * 256;
    float v[8], s = 0.f;
    #pragma unroll
    for (int i = 0; i < 8; ++i) { v[i] = p[lane + i * 32]; s += v[i]; }
    #pragma unroll
    for (int m = 16; m > 0; m >>= 1) s += __shfl_xor(s, m, 32);
    const float mean = s * (1.0f / 256.0f);
    float q = 0.f;
    #pragma unroll
    for (int i = 0; i < 8; ++i) { const float d0 = v[i] - mean; q += d0 * d0; }
    #pragma unroll
    for (int m = 16; m > 0; m >>= 1) q += __shfl_xor(q, m, 32);
    const float inv = rsqrtf(q * (1.0f / 256.0f) + 1e-6f);
    #pragma unroll
    for (int i = 0; i < 8; ++i) {
        const int c = lane + i * 32;
        out[(size_t)row * 256 + c] = g[c] * (v[i] - mean) * inv + b[c];
    }
}

// ---------------------------------------------------------------------------
// Token-level MHA core: softmax(QK^T/sqrt(dk) + keymask) V  per (bn, h, q).
// Grid = 96 blocks (one per line), block = 384 threads = (h, q) pairs.
// ---------------------------------------------------------------------------
__global__ void token_attn_kernel(const float* __restrict__ q, const float* __restrict__ k,
                                  const float* __restrict__ v, const int* __restrict__ codes,
                                  float* __restrict__ ctx) {
    const int bn = blockIdx.x, tid = threadIdx.x;
    __shared__ float km[LL];
    if (tid < LL) km[tid] = (codes[bn * LL + tid] > 0) ? 0.0f : NEGBIG;
    __syncthreads();
    const int h = tid / LL, qi = tid % LL;
    const float* qp = q + ((size_t)(bn * LL + qi)) * 256 + h * DK;
    float qr[DK];
    #pragma unroll
    for (int c = 0; c < DK; c += 4) {
        const float4 t = *(const float4*)(qp + c);
        qr[c] = t.x; qr[c+1] = t.y; qr[c+2] = t.z; qr[c+3] = t.w;
    }
    float s[LL], mx = -3.4e38f;
    #pragma unroll
    for (int kk = 0; kk < LL; ++kk) {
        const float* kp = k + ((size_t)(bn * LL + kk)) * 256 + h * DK;
        float d0 = 0.f;
        #pragma unroll
        for (int c = 0; c < DK; c += 4) {
            const float4 t = *(const float4*)(kp + c);
            d0 += qr[c]*t.x + qr[c+1]*t.y + qr[c+2]*t.z + qr[c+3]*t.w;
        }
        d0 = d0 * RSQRT_DK + km[kk];
        s[kk] = d0; mx = fmaxf(mx, d0);
    }
    float den = 0.f;
    #pragma unroll
    for (int kk = 0; kk < LL; ++kk) { const float e = __expf(s[kk] - mx); s[kk] = e; den += e; }
    const float inv = 1.0f / den;
    float acc[DK];
    #pragma unroll
    for (int c = 0; c < DK; ++c) acc[c] = 0.f;
    #pragma unroll
    for (int kk = 0; kk < LL; ++kk) {
        const float w0 = s[kk] * inv;
        const float* vp = v + ((size_t)(bn * LL + kk)) * 256 + h * DK;
        #pragma unroll
        for (int c = 0; c < DK; c += 4) {
            const float4 t = *(const float4*)(vp + c);
            acc[c] += w0*t.x; acc[c+1] += w0*t.y; acc[c+2] += w0*t.z; acc[c+3] += w0*t.w;
        }
    }
    float* op = ctx + ((size_t)(bn * LL + qi)) * 256 + h * DK;
    #pragma unroll
    for (int c = 0; c < DK; ++c) op[c] = acc[c];
}

// ---------------------------------------------------------------------------
// Line-to-line attention with the j-sum folded in:
//   ctxsum[b,i,q,h,:] = sum_j lm[b,j] * softmax_k(Q_i K_j^T + mask_j) V_j
// Grid = 96 blocks (b,i); block = 384 threads = (h, q).
// ---------------------------------------------------------------------------
__global__ void line_attn_kernel(const float* __restrict__ Q, const float* __restrict__ K,
                                 const float* __restrict__ V, const int* __restrict__ codes,
                                 float* __restrict__ ctx) {
    const int bi = blockIdx.x;
    const int b = bi / NN, i = bi % NN;
    const int tid = threadIdx.x;
    const int h = tid / LL, qi = tid % LL;
    __shared__ float kmj[LL];
    __shared__ float lmj;

    const float* qp = Q + ((size_t)((b * NN + i) * LL + qi)) * 256 + h * DK;
    float qr[DK];
    #pragma unroll
    for (int c = 0; c < DK; c += 4) {
        const float4 t = *(const float4*)(qp + c);
        qr[c] = t.x; qr[c+1] = t.y; qr[c+2] = t.z; qr[c+3] = t.w;
    }
    float acc[DK];
    #pragma unroll
    for (int c = 0; c < DK; ++c) acc[c] = 0.f;

    for (int j = 0; j < NN; ++j) {
        __syncthreads();
        if (tid < LL) kmj[tid] = (codes[(b * NN + j) * LL + tid] > 0) ? 0.0f : NEGBIG;
        if (tid == 0) {
            int any = 0;
            for (int t = 0; t < LL; ++t) any |= (codes[(b * NN + j) * LL + t] > 0);
            lmj = any ? 1.0f : 0.0f;
        }
        __syncthreads();
        const float lm = lmj;
        if (lm == 0.0f) continue;

        float s[LL], mx = -3.4e38f;
        #pragma unroll
        for (int kk = 0; kk < LL; ++kk) {
            const float* kp = K + ((size_t)((b * NN + j) * LL + kk)) * 256 + h * DK;
            float d0 = 0.f;
            #pragma unroll
            for (int c = 0; c < DK; c += 4) {
                const float4 t = *(const float4*)(kp + c);
                d0 += qr[c]*t.x + qr[c+1]*t.y + qr[c+2]*t.z + qr[c+3]*t.w;
            }
            d0 = d0 * RSQRT_DK + kmj[kk];
            s[kk] = d0; mx = fmaxf(mx, d0);
        }
        float den = 0.f;
        #pragma unroll
        for (int kk = 0; kk < LL; ++kk) { const float e = __expf(s[kk] - mx); s[kk] = e; den += e; }
        const float winv = lm / den;
        #pragma unroll
        for (int kk = 0; kk < LL; ++kk) {
            const float w0 = s[kk] * winv;
            const float* vp = V + ((size_t)((b * NN + j) * LL + kk)) * 256 + h * DK;
            #pragma unroll
            for (int c = 0; c < DK; c += 4) {
                const float4 t = *(const float4*)(vp + c);
                acc[c] += w0*t.x; acc[c+1] += w0*t.y; acc[c+2] += w0*t.z; acc[c+3] += w0*t.w;
            }
        }
    }
    float* op = ctx + ((size_t)((b * NN + i) * LL + qi)) * 256 + h * DK;
    #pragma unroll
    for (int c = 0; c < DK; ++c) op[c] = acc[c];
}

// per-batch sum of linemask (for the folded bias term)
__global__ void lmsum_kernel(const int* __restrict__ codes, float* __restrict__ lmsum) {
    __shared__ float lm[BN];
    const int tid = threadIdx.x;
    if (tid < BN) {
        int any = 0;
        for (int t = 0; t < LL; ++t) any |= (codes[tid * LL + t] > 0);
        lm[tid] = any ? 1.0f : 0.0f;
    }
    __syncthreads();
    if (tid < BB) {
        float s = 0.f;
        for (int j = 0; j < NN; ++j) s += lm[tid * NN + j];
        lmsum[tid] = s;
    }
}

// final = y + enc; write transposed [B, L, N, D], plus mask as floats after it.
__global__ void final_out_kernel(const float* __restrict__ y, const float* __restrict__ enc,
                                 const int* __restrict__ codes, float* __restrict__ out) {
    const int gid = blockIdx.x * 256 + threadIdx.x;       // over ROWS*256
    const int d = gid & 255, r = gid >> 8;
    const int l = r % LL, n = (r / LL) % NN, b = r / (LL * NN);
    out[(((size_t)b * LL + l) * NN + n) * 256 + d] = y[gid] + enc[gid];
    if (d == 0)
        out[(size_t)ROWS * 256 + (size_t)(b * NN + n) * LL + l] = (codes[r] > 0) ? 1.0f : 0.0f;
}

// ---------------------------------------------------------------------------
// Host-side orchestration
// ---------------------------------------------------------------------------
extern "C" void kernel_launch(void* const* d_in, const int* in_sizes, int n_in,
                              void* d_out, int out_size, void* d_ws, size_t ws_size,
                              hipStream_t stream) {
    const int*   inputCode     = (const int*)  d_in[0];
    const int*   inputCodeChar = (const int*)  d_in[1];
    const float* char_emb      = (const float*)d_in[2];
    const float* conv_w        = (const float*)d_in[3];
    const float* conv_b        = (const float*)d_in[4];
    const float* token_emb     = (const float*)d_in[5];
    const float* blk_Wq = (const float*)d_in[6],  *blk_bq = (const float*)d_in[7];
    const float* blk_Wk = (const float*)d_in[8],  *blk_bk = (const float*)d_in[9];
    const float* blk_Wv = (const float*)d_in[10], *blk_bv = (const float*)d_in[11];
    const float* blk_Wo = (const float*)d_in[12], *blk_bo = (const float*)d_in[13];
    const float* blk_ln1_g = (const float*)d_in[14], *blk_ln1_b = (const float*)d_in[15];
    const float* blk_W1 = (const float*)d_in[16], *blk_b1 = (const float*)d_in[17];
    const float* blk_W2 = (const float*)d_in[18], *blk_b2 = (const float*)d_in[19];
    const float* blk_ln2_g = (const float*)d_in[20], *blk_ln2_b = (const float*)d_in[21];
    const float* la_Wq = (const float*)d_in[22], *la_bq = (const float*)d_in[23];
    const float* la_Wk = (const float*)d_in[24], *la_bk = (const float*)d_in[25];
    const float* la_Wv = (const float*)d_in[26], *la_bv = (const float*)d_in[27];
    const float* la_Wo = (const float*)d_in[28], *la_bo = (const float*)d_in[29];
    const float* lf_W1 = (const float*)d_in[30], *lf_b1 = (const float*)d_in[31];
    const float* lf_W2 = (const float*)d_in[32], *lf_b2 = (const float*)d_in[33];
    const float* ln_g  = (const float*)d_in[34], *ln_b  = (const float*)d_in[35];

    const size_t SZ = (size_t)ROWS * D_MODEL;   // 1,179,648 floats
    float* w0     = (float*)d_ws;
    float* charEm = w0;
    float* bufA   = w0 + 1 * SZ;   // residual carry (x)
    float* bufB   = w0 + 2 * SZ;   // working (a)
    float* hbuf   = w0 + 3 * SZ;
    float* qbuf   = w0 + 4 * SZ;
    float* kbuf   = w0 + 5 * SZ;
    float* vbuf   = w0 + 6 * SZ;
    float* cbuf   = w0 + 7 * SZ;
    float* ffn    = w0 + 8 * SZ;                       // 4608*1024
    float* lmsum  = w0 + 8 * SZ + (size_t)ROWS * 1024;

    const dim3 blk256(256);
    const dim3 gNN(ROWS / 64, 256 / 64);     // GEMM N=256  -> (72, 4)
    const dim3 gN4(ROWS / 64, 1024 / 64);    // GEMM N=1024 -> (72, 16)
    const dim3 gCC(ROWS / 32, 256 / 64);     // char-conv   -> (144, 4)
    const int  gElem = (int)(SZ / 256);      // 4608
    const int  gLN   = ROWS / 8;             // 576

    // Embedding stage
    embed_kernel<<<gElem, blk256, 0, stream>>>(inputCode, token_emb, bufA);
    charconv_wmma_kernel<<<gCC, blk256, 0, stream>>>(inputCodeChar, char_emb, conv_w, conv_b, charEm);

    float* X = bufA;   // carry
    float* A = bufB;   // working
    for (int t = 0; t < 20; ++t) {
        const float* Wq = blk_Wq + (size_t)t * 65536, *bq = blk_bq + t * 256;
        const float* Wk = blk_Wk + (size_t)t * 65536, *bk = blk_bk + t * 256;
        const float* Wv = blk_Wv + (size_t)t * 65536, *bv = blk_bv + t * 256;
        const float* Wo = blk_Wo + (size_t)t * 65536, *bo = blk_bo + t * 256;
        const float* W1 = blk_W1 + (size_t)t * 256 * 1024, *b1 = blk_b1 + t * 1024;
        const float* W2 = blk_W2 + (size_t)t * 1024 * 256, *b2 = blk_b2 + t * 256;

        // a = x + charEm; h = LN1(a)   (fused)
        add_ln_kernel<<<gLN, blk256, 0, stream>>>(X, charEm, blk_ln1_g + t * 256,
                                                  blk_ln1_b + t * 256, A, hbuf);
        gemm_wmma_kernel<<<gNN, blk256, 0, stream>>>(hbuf, Wq, bq, nullptr, nullptr, qbuf, ROWS, 256, 256, 0);
        gemm_wmma_kernel<<<gNN, blk256, 0, stream>>>(hbuf, Wk, bk, nullptr, nullptr, kbuf, ROWS, 256, 256, 0);
        gemm_wmma_kernel<<<gNN, blk256, 0, stream>>>(hbuf, Wv, bv, nullptr, nullptr, vbuf, ROWS, 256, 256, 0);
        token_attn_kernel<<<BN, 384, 0, stream>>>(qbuf, kbuf, vbuf, inputCode, cbuf);
        gemm_wmma_kernel<<<gNN, blk256, 0, stream>>>(cbuf, Wo, bo, A, nullptr, A, ROWS, 256, 256, 0);
        ln_kernel<<<gLN, blk256, 0, stream>>>(A, blk_ln2_g + t * 256, blk_ln2_b + t * 256, hbuf);
        gemm_wmma_kernel<<<gN4, blk256, 0, stream>>>(hbuf, W1, b1, nullptr, nullptr, ffn, ROWS, 256, 1024, 1);
        gemm_wmma_kernel<<<gNN, blk256, 0, stream>>>(ffn, W2, b2, A, nullptr, A, ROWS, 1024, 256, 0);
        float* tmp = X; X = A; A = tmp;                                          // x_{t+1} = a
    }
    float* enc = X;          // encoder output
    float* work = A;         // free buffer

    // Line-to-line attention stage
    gemm_wmma_kernel<<<gNN, blk256, 0, stream>>>(enc, la_Wq, la_bq, nullptr, nullptr, qbuf, ROWS, 256, 256, 0);
    gemm_wmma_kernel<<<gNN, blk256, 0, stream>>>(enc, la_Wk, la_bk, nullptr, nullptr, kbuf, ROWS, 256, 256, 0);
    gemm_wmma_kernel<<<gNN, blk256, 0, stream>>>(enc, la_Wv, la_bv, nullptr, nullptr, vbuf, ROWS, 256, 256, 0);
    lmsum_kernel<<<1, 128, 0, stream>>>(inputCode, lmsum);
    line_attn_kernel<<<BN, 384, 0, stream>>>(qbuf, kbuf, vbuf, inputCode, cbuf);
    // summed = ctxsum @ Wo + lmsum[b] * bo
    gemm_wmma_kernel<<<gNN, blk256, 0, stream>>>(cbuf, la_Wo, la_bo, nullptr, lmsum, work, ROWS, 256, 256, 0);
    ln_kernel<<<gLN, blk256, 0, stream>>>(work, ln_g, ln_b, hbuf);
    gemm_wmma_kernel<<<gN4, blk256, 0, stream>>>(hbuf, lf_W1, lf_b1, nullptr, nullptr, ffn, ROWS, 256, 1024, 1);
    gemm_wmma_kernel<<<gNN, blk256, 0, stream>>>(ffn, lf_W2, lf_b2, work, nullptr, work, ROWS, 1024, 256, 0);
    final_out_kernel<<<gElem, blk256, 0, stream>>>(work, enc, inputCode, (float*)d_out);
}
